// RNN_7988639171178
// MI455X (gfx1250) — compile-verified, matrix operand order checked by reading, not run on hardware
//
#include <hip/hip_runtime.h>

// ---------------------------------------------------------------------------
// RNN (returns last-timestep output only) for MI455X / gfx1250.
//
//   W_i2h = [W_x | W_h]  (cols 0..IN-1, IN..IN+HID-1)
//   Z[t]    = x_t @ W_x.T + b_i2h        -> one big WMMA bf16 GEMM (34 GFLOP)
//   c_{t+1} = sigmoid(Z[t] + W_h @ c_t)  -> persistent kernel, W_h LDS-resident
//                                           (320 KB WGP LDS), grid barrier/step
//   out     = [x_last, c_{S-1}] @ W_i2o.T + b_i2o
// ---------------------------------------------------------------------------

#define SEQ  2048
#define IN   2048
#define HID  4096
#define OUT  2048
#define KW   (IN + HID)     // 6144, row stride of W_i2h / W_i2o

#define RNN_BLOCKS 128      // persistent blocks; 32 hidden rows each
#define ROWS_PB    (HID / RNN_BLOCKS)            // 32
#define LDS_W_BYTES (ROWS_PB * HID * 2)          // 256 KB bf16 weights
#define LDS_H_BYTES (HID * 4)                    // 16 KB f32 hidden
#define LDS_BYTES   (LDS_W_BYTES + LDS_H_BYTES)  // 272 KB < 320 KB WGP LDS

typedef __attribute__((ext_vector_type(16))) __bf16 v16bf;
typedef __attribute__((ext_vector_type(8)))  __bf16 v8bf;
typedef __attribute__((ext_vector_type(8)))  float  v8f;

// ---- fp32 -> bf16 (round to nearest even) ---------------------------------
__device__ __forceinline__ unsigned short f2bf(float f) {
    union { float f; unsigned u; } v; v.f = f;
    unsigned r = v.u + 0x7FFFu + ((v.u >> 16) & 1u);
    return (unsigned short)(r >> 16);
}

__global__ __launch_bounds__(256) void k_f32_to_bf16(const float* __restrict__ src,
                                                     unsigned short* __restrict__ dst,
                                                     int n) {
    int i = (blockIdx.x * 256 + threadIdx.x) * 4;
    if (i + 3 < n) {
        float4 v = *(const float4*)(src + i);
        ushort4 o;
        o.x = f2bf(v.x); o.y = f2bf(v.y); o.z = f2bf(v.z); o.w = f2bf(v.w);
        *(ushort4*)(dst + i) = o;
    }
}

__global__ __launch_bounds__(256) void k_zero(unsigned* __restrict__ p, int n) {
    int i = blockIdx.x * 256 + threadIdx.x;
    if (i < n) p[i] = 0u;
}

// ---- 16x32 bf16 fragment load (ISA 7.12.2 layout) -------------------------
__device__ __forceinline__ v16bf load_frag_16x32(const __bf16* __restrict__ base,
                                                 int ld, int row0, int k0, int lane) {
    const __bf16* p = base + (size_t)(row0 + (lane & 15)) * ld + k0 + ((lane >> 4) << 3);
    v8bf lo = *(const v8bf*)(p);        // global_load_b128
    v8bf hi = *(const v8bf*)(p + 16);   // global_load_b128
    v16bf f;
#pragma unroll
    for (int i = 0; i < 8; ++i) { f[i] = lo[i]; f[i + 8] = hi[i]; }
    return f;
}

// ---- WMMA GEMM: Z[m,n] = sum_k X[m,k] * W_i2h[n,k] + b_i2h[n] -------------
__global__ __launch_bounds__(256) void k_gemm_z(const __bf16* __restrict__ Xb,
                                                const __bf16* __restrict__ Wb,
                                                const float* __restrict__ bias,
                                                float* __restrict__ Z) {
    const int lane = threadIdx.x & 31;
    const int wave = threadIdx.x >> 5;
    const int m0 = blockIdx.x * 128 + wave * 16;
    const int n0 = blockIdx.y * 64;

    v8f acc0 = {}, acc1 = {}, acc2 = {}, acc3 = {};

    for (int k = 0; k < IN; k += 32) {
        v16bf a  = load_frag_16x32(Xb, IN, m0, k, lane);
        v16bf b0 = load_frag_16x32(Wb, KW, n0 +  0, k, lane);
        v16bf b1 = load_frag_16x32(Wb, KW, n0 + 16, k, lane);
        v16bf b2 = load_frag_16x32(Wb, KW, n0 + 32, k, lane);
        v16bf b3 = load_frag_16x32(Wb, KW, n0 + 48, k, lane);
        acc0 = __builtin_amdgcn_wmma_f32_16x16x32_bf16(false, a, false, b0, (short)0, acc0, false, false);
        acc1 = __builtin_amdgcn_wmma_f32_16x16x32_bf16(false, a, false, b1, (short)0, acc1, false, false);
        acc2 = __builtin_amdgcn_wmma_f32_16x16x32_bf16(false, a, false, b2, (short)0, acc2, false, false);
        acc3 = __builtin_amdgcn_wmma_f32_16x16x32_bf16(false, a, false, b3, (short)0, acc3, false, false);
    }

    const int nlo = lane & 15;
    const int mhi = (lane >> 4) * 8;
    v8f accs[4] = {acc0, acc1, acc2, acc3};
#pragma unroll
    for (int j = 0; j < 4; ++j) {
        int n = n0 + 16 * j + nlo;
        float bv = bias[n];
#pragma unroll
        for (int r = 0; r < 8; ++r) {
            Z[(size_t)(m0 + mhi + r) * HID + n] = accs[j][r] + bv;
        }
    }
}

// ---- Grid-wide generation barrier (device-scope atomics in d_ws) ----------
__device__ __forceinline__ void grid_sync(unsigned* count, unsigned* gen, unsigned nblk) {
    __threadfence();     // make this step's h stores visible device-wide
    __syncthreads();
    if (threadIdx.x == 0) {
        unsigned g = __hip_atomic_load(gen, __ATOMIC_RELAXED, __HIP_MEMORY_SCOPE_AGENT);
        unsigned prev = __hip_atomic_fetch_add(count, 1u, __ATOMIC_ACQ_REL, __HIP_MEMORY_SCOPE_AGENT);
        if (prev == nblk - 1) {
            __hip_atomic_store(count, 0u, __ATOMIC_RELAXED, __HIP_MEMORY_SCOPE_AGENT);
            __hip_atomic_fetch_add(gen, 1u, __ATOMIC_RELEASE, __HIP_MEMORY_SCOPE_AGENT);
        } else {
            while (__hip_atomic_load(gen, __ATOMIC_ACQUIRE, __HIP_MEMORY_SCOPE_AGENT) == g) {
                __builtin_amdgcn_s_sleep(2);
            }
        }
    }
    __syncthreads();
}

// ---- Persistent recurrence: c_{t+1} = sigmoid(Z[t] + W_h @ c_t) -----------
// 128 blocks, one per WGP (forced by 272 KB LDS). Each block owns 32 hidden
// rows; its bf16 weight slice lives in LDS for all 2047 steps. Per step only
// the 16 KB hidden vector moves through L2.
__global__ __launch_bounds__(256) void k_rnn_persistent(const __bf16* __restrict__ Wb,
                                                        const float* __restrict__ Z,
                                                        float* __restrict__ h0,
                                                        float* __restrict__ h1,
                                                        unsigned* __restrict__ count,
                                                        unsigned* __restrict__ gen) {
    extern __shared__ char smem[];
    __bf16* w_lds = (__bf16*)smem;                       // 32 x 4096 bf16
    float*  h_lds = (float*)(smem + LDS_W_BYTES);        // 4096 f32

    const int tid   = threadIdx.x;
    const int row_l = tid >> 3;                          // 0..31
    const int part  = tid & 7;                           // 0..7
    const int row_g = blockIdx.x * ROWS_PB + row_l;

    // One-time fill: 32 rows x 4096 bf16 (hidden part of W_i2h) into LDS.
    for (int i = tid * 8; i < ROWS_PB * HID; i += 256 * 8) {
        int r = i >> 12;          // / HID
        int c = i & (HID - 1);
        *(v8bf*)(w_lds + i) =
            *(const v8bf*)(Wb + (size_t)(blockIdx.x * ROWS_PB + r) * KW + IN + c);
    }
    __syncthreads();

    const __bf16* wr = w_lds + row_l * HID + part * 512;
    const float*  hp = h_lds + part * 512;

    for (int t = 0; t < SEQ - 1; ++t) {
        const float* hin  = (t & 1) ? h1 : h0;
        float*       hout = (t & 1) ? h0 : h1;

        // Stage broadcast hidden vector into LDS (16 floats / thread).
        for (int i = tid * 4; i < HID; i += 256 * 4) {
            *(float4*)(h_lds + i) = *(const float4*)(hin + i);
        }
        __syncthreads();

        // 512-element partial dot from LDS (ds_load_b128 weight + hidden).
        float s = 0.0f;
#pragma unroll 4
        for (int k = 0; k < 512; k += 8) {
            v8bf w = *(const v8bf*)(wr + k);
            float4 ha = *(const float4*)(hp + k);
            float4 hb = *(const float4*)(hp + k + 4);
            s += (float)w[0] * ha.x + (float)w[1] * ha.y +
                 (float)w[2] * ha.z + (float)w[3] * ha.w;
            s += (float)w[4] * hb.x + (float)w[5] * hb.y +
                 (float)w[6] * hb.z + (float)w[7] * hb.w;
        }
        s += __shfl_down(s, 4, 32);
        s += __shfl_down(s, 2, 32);
        s += __shfl_down(s, 1, 32);
        if (part == 0) {
            float z = s + Z[(size_t)t * HID + row_g];
            hout[row_g] = 1.0f / (1.0f + __expf(-z));    // v_exp_f32
        }
        grid_sync(count, gen, (unsigned)gridDim.x);
    }
}

// ---- Final projection: out = [x_last, h] @ W_i2o.T + b_i2o ----------------
__global__ __launch_bounds__(256) void k_final(const float* __restrict__ x_last,
                                               const float* __restrict__ Wo,
                                               const float* __restrict__ bo,
                                               const float* __restrict__ h,
                                               float* __restrict__ out) {
    const int row  = blockIdx.x * 32 + (threadIdx.x >> 3);
    const int part = threadIdx.x & 7;
    const float* wr = Wo + (size_t)row * KW;

    float s = 0.0f;
    const int base = part * 768;               // 6144 / 8 lanes
#pragma unroll 4
    for (int i = 0; i < 768; i += 4) {
        int k = base + i;
        float4 w = *(const float4*)(wr + k);
        float v0, v1, v2, v3;
        if (k + 3 < IN) {
            float4 x = *(const float4*)(x_last + k);
            v0 = x.x; v1 = x.y; v2 = x.z; v3 = x.w;
        } else {
            float4 hh = *(const float4*)(h + (k - IN));
            v0 = hh.x; v1 = hh.y; v2 = hh.z; v3 = hh.w;
        }
        s += w.x * v0 + w.y * v1 + w.z * v2 + w.w * v3;
    }
    s += __shfl_down(s, 4, 32);
    s += __shfl_down(s, 2, 32);
    s += __shfl_down(s, 1, 32);
    if (part == 0) out[row] = s + bo[row];
}

// ---------------------------------------------------------------------------
extern "C" void kernel_launch(void* const* d_in, const int* in_sizes, int n_in,
                              void* d_out, int out_size, void* d_ws, size_t ws_size,
                              hipStream_t stream) {
    const float* input = (const float*)d_in[0];   // [SEQ, 1, IN]
    const float* W_i2h = (const float*)d_in[1];   // [HID, IN+HID]
    const float* b_i2h = (const float*)d_in[2];   // [HID]
    const float* W_i2o = (const float*)d_in[3];   // [OUT, IN+HID]
    const float* b_i2o = (const float*)d_in[4];   // [OUT]
    float* out = (float*)d_out;

    // Workspace layout (bytes):
    //   [0, 50331648)          Wbf  : bf16 W_i2h (HID x KW)
    //   [50331648, 58720256)   Xbf  : bf16 input (SEQ x IN)
    //   [58720256, 92274688)   Z    : f32  (SEQ x HID)
    //   [92274688, +16K)       h0, then h1 (+16K), then barrier ctrl (+256B)
    char* ws = (char*)d_ws;
    unsigned short* Wbf_u = (unsigned short*)(ws);
    unsigned short* Xbf_u = (unsigned short*)(ws + 50331648ull);
    float*          Z     = (float*)(ws + 58720256ull);
    float*          h0    = (float*)(ws + 92274688ull);
    float*          h1    = (float*)(ws + 92274688ull + 16384ull);
    unsigned*       ctrl  = (unsigned*)(ws + 92274688ull + 32768ull);   // [count, gen]
    const __bf16*   Wbf   = (const __bf16*)Wbf_u;
    const __bf16*   Xbf   = (const __bf16*)Xbf_u;

    // 1) Precision conversion (one HBM pass) + state/ctrl init.
    {
        int nW = HID * KW;                        // 25,165,824
        k_f32_to_bf16<<<nW / 1024, 256, 0, stream>>>(W_i2h, Wbf_u, nW);
        int nX = SEQ * IN;                        // 4,194,304
        k_f32_to_bf16<<<nX / 1024, 256, 0, stream>>>(input, Xbf_u, nX);
        k_zero<<<HID / 256, 256, 0, stream>>>((unsigned*)h0, HID);
        k_zero<<<1, 256, 0, stream>>>(ctrl, 64);
    }

    // 2) Batch-parallel part: Z = X @ W_x.T + b_i2h via WMMA bf16.
    {
        dim3 grid(SEQ / 128, HID / 64);
        k_gemm_z<<<grid, 256, 0, stream>>>(Xbf, Wbf, b_i2h, Z);
    }

    // 3) Serial recurrence, single persistent kernel (2047 grid barriers).
    //    Double-buffered h: step t reads buf[t&1], writes buf[(t+1)&1];
    //    after 2047 steps the carry entering the last timestep is in h1.
    k_rnn_persistent<<<RNN_BLOCKS, 256, LDS_BYTES, stream>>>(Wbf, Z, h0, h1,
                                                             ctrl + 0, ctrl + 1);

    // 4) Final timestep output projection (fp32, single pass over W_i2o).
    {
        const float* x_last = input + (size_t)(SEQ - 1) * IN;
        k_final<<<OUT / 32, 256, 0, stream>>>(x_last, W_i2o, b_i2o, h1, out);
    }
}
